// GGNN_71124658422333
// MI455X (gfx1250) — matrix operand order repaired and checked by reading, not compiled, not run on hardware
//
#include <hip/hip_runtime.h>
#include <hip/hip_bf16.h>
#include <math.h>

typedef __attribute__((ext_vector_type(16))) _Float16 v16h;
typedef __attribute__((ext_vector_type(8)))  float    v8f;

static __device__ __forceinline__ unsigned int pack2h(float lo, float hi) {
    _Float16 l = (_Float16)lo, h = (_Float16)hi;
    unsigned short ls = __builtin_bit_cast(unsigned short, l);
    unsigned short hs = __builtin_bit_cast(unsigned short, h);
    return (unsigned int)ls | ((unsigned int)hs << 16);
}

__global__ void zero_kernel(float* __restrict__ p, int n) {
    int i = blockIdx.x * blockDim.x + threadIdx.x;
    const int stride = gridDim.x * blockDim.x;
    for (; i < n; i += stride) p[i] = 0.0f;
}

// Pre-convert W (f32 [NOUT x K] row-major) into f16, swizzled into this GEMM's
// per-lane B-fragment order (contiguous k, matching the A staging):
//   dword t = ((nt*KT + kt)*32 + lane)*8 + j  holds {W[n][k], W[n][k+1]}
//   with n = nt*16 + (lane&15), k = kt*32 + 16*(lane>>4) + 2*j.
// The GEMM then loads B fragments with two b128 loads and zero VALU.
template<int K, int NOUT>
__global__ void pack_weights(const float* __restrict__ W,
                             unsigned int* __restrict__ Wp) {
    constexpr int KT  = K / 32;
    constexpr int TOT = NOUT * K / 2;   // total packed dwords
    const int t = blockIdx.x * blockDim.x + threadIdx.x;
    if (t >= TOT) return;
    const int j    = t & 7;
    const int lane = (t >> 3) & 31;
    const int g    = t >> 8;            // (nt*KT + kt)
    const int kt   = g % KT;
    const int nt   = g / KT;
    const int n    = nt * 16 + (lane & 15);
    const int k    = kt * 32 + 16 * (lane >> 4) + 2 * j;
    Wp[t] = pack2h(W[(size_t)n * K + k], W[(size_t)n * K + k + 1]);
}

// D[M x NOUT] = act(A[M x K]) * W^T + bias. A is f32 (opt. relu fused on load),
// weights pre-packed f16 fragments. One 16-row M tile per block; wave w handles
// output columns [16w, 16w+16).
template<int K, int NOUT, bool RELU>
__global__ void gemm_wmma(const float* __restrict__ A,
                          const unsigned int* __restrict__ Wp,
                          const float* __restrict__ bias,
                          float* __restrict__ D,
                          int M) {
    constexpr int NT   = NOUT / 16;      // waves per block
    constexpr int KT   = K / 32;         // wmma K-steps
    constexpr int ROWU = K / 2 + 4;      // LDS row stride in dwords (padded)
    __shared__ unsigned int ldsA[16 * ROWU];

    const int tid   = threadIdx.x;
    const int lane  = tid & 31;
    const int wv    = tid >> 5;          // N-tile index for this wave
    const int half  = lane >> 4;
    const int l15   = lane & 15;
    const int row0  = blockIdx.x * 16;

    // ---- Stage 16 x K A-tile into LDS as packed f16 pairs (convert + opt relu) ----
    constexpr int TOTU    = 16 * (K / 2);
    constexpr int THREADS = NT * 32;
    constexpr int ITER    = TOTU / THREADS;
    #pragma unroll
    for (int t = 0; t < ITER; ++t) {
        const int idx  = tid + t * THREADS;
        const int row  = idx / (K / 2);
        const int col2 = idx % (K / 2);
        float vx = 0.0f, vy = 0.0f;
        const int gr = row0 + row;
        if (gr < M) {
            const float* p = A + (size_t)gr * K + 2 * col2;
            vx = p[0]; vy = p[1];
            if (RELU) { vx = fmaxf(vx, 0.0f); vy = fmaxf(vy, 0.0f); }
        }
        ldsA[row * ROWU + col2] = pack2h(vx, vy);
    }
    __syncthreads();

    v8f acc = {0.f, 0.f, 0.f, 0.f, 0.f, 0.f, 0.f, 0.f};
    const int ncol0 = wv * 16;
    // This wave+lane's pre-packed B fragments: 8 consecutive dwords per K-step,
    // stride 256 dwords between K-steps. 32B-aligned -> 2x global_load_b128.
    const unsigned int* wp = Wp + (size_t)wv * KT * 256 + lane * 8;

    #pragma unroll
    for (int kt = 0; kt < KT; ++kt) {
        const int K0 = kt * 32;
        // A fragment (16x32 f16): lanes 0-15: K 0-15 (V0-7);
        // lanes 16-31: K 16-31. Consecutive-dword run -> ds_load_b128.
        unsigned int au[8];
        #pragma unroll
        for (int i = 0; i < 8; ++i) {
            const int uidx = (K0 >> 1) + 8 * half + i;
            au[i] = ldsA[l15 * ROWU + uidx];
        }
        // B fragment: direct vector loads from pre-packed weights.
        const uint4 b0 = *(const uint4*)(wp + (size_t)kt * 256);
        const uint4 b1 = *(const uint4*)(wp + (size_t)kt * 256 + 4);
        unsigned int bu[8] = {b0.x, b0.y, b0.z, b0.w, b1.x, b1.y, b1.z, b1.w};
        v16h a, b;
        __builtin_memcpy(&a, au, 32);
        __builtin_memcpy(&b, bu, 32);
        acc = __builtin_amdgcn_wmma_f32_16x16x32_f16(false, a, false, b,
                                                     (short)0, acc, false, false);
    }

    // D layout: VGPR r -> M = r + 8*half, N = lane&15
    float accs[8];
    __builtin_memcpy(accs, &acc, 32);
    const float bv = bias[ncol0 + l15];
    #pragma unroll
    for (int r = 0; r < 8; ++r) {
        const int gm = row0 + r + 8 * half;
        if (gm < M) D[(size_t)gm * NOUT + ncol0 + l15] = accs[r] + bv;
    }
}

// One wave per edge: gather row h[src], atomic-add into agg[dst].
template<int C>
__global__ void scatter_add(const float* __restrict__ h,
                            const int* __restrict__ src,
                            const int* __restrict__ dst,
                            float* __restrict__ agg, int E) {
    const int gid  = blockIdx.x * blockDim.x + threadIdx.x;
    const int e    = gid >> 5;
    const int lane = gid & 31;
    if (e >= E) return;
    const int s = src[e];
    const int d = dst[e];
    constexpr int VPT = C / 32;   // floats per lane (4 or 2)
    const float* hp = h   + (size_t)s * C + lane * VPT;
    float*       ap = agg + (size_t)d * C + lane * VPT;
    if constexpr (VPT == 4) {
        const float4 v = *(const float4*)hp;
        atomicAdd(ap + 0, v.x);
        atomicAdd(ap + 1, v.y);
        atomicAdd(ap + 2, v.z);
        atomicAdd(ap + 3, v.w);
    } else {
        const float2 v = *(const float2*)hp;
        atomicAdd(ap + 0, v.x);
        atomicAdd(ap + 1, v.y);
    }
}

// One wave per row of 64: log_softmax via wave32 shuffle reductions.
__global__ void log_softmax64(const float* __restrict__ in,
                              float* __restrict__ out, int M) {
    const int gid  = blockIdx.x * blockDim.x + threadIdx.x;
    const int row  = gid >> 5;
    const int lane = gid & 31;
    if (row >= M) return;
    float2 v = *(const float2*)(in + (size_t)row * 64 + lane * 2);
    float m = fmaxf(v.x, v.y);
    #pragma unroll
    for (int off = 16; off > 0; off >>= 1) m = fmaxf(m, __shfl_xor(m, off, 32));
    float s = __expf(v.x - m) + __expf(v.y - m);
    #pragma unroll
    for (int off = 16; off > 0; off >>= 1) s += __shfl_xor(s, off, 32);
    const float lse = m + __logf(s);
    float2 o;
    o.x = v.x - lse;
    o.y = v.y - lse;
    *(float2*)(out + (size_t)row * 64 + lane * 2) = o;
}

extern "C" void kernel_launch(void* const* d_in, const int* in_sizes, int n_in,
                              void* d_out, int out_size, void* d_ws, size_t ws_size,
                              hipStream_t stream) {
    const float* x  = (const float*)d_in[0];   // [N, 256]
    const int*   ei = (const int*)d_in[1];     // [2, E]
    const float* W1 = (const float*)d_in[2];   // [128, 256]
    const float* b1 = (const float*)d_in[3];   // [128]
    const float* W2 = (const float*)d_in[4];   // [64, 128]
    const float* b2 = (const float*)d_in[5];   // [64]
    float* out = (float*)d_out;                // [N, 64] f32

    const int N = in_sizes[0] / 256;
    const int E = in_sizes[1] / 2;
    const int* src = ei;
    const int* dst = ei + E;

    // Workspace: h (N x 128 f32, reused as N x 64), agg (N x 128 f32, reused as
    // N x 64), then packed f16 weights. Total ~51.3 MB.
    float* h   = (float*)d_ws;
    float* agg = h + (size_t)N * 128;
    unsigned int* W1p = (unsigned int*)(agg + (size_t)N * 128);  // 128*256/2 = 16384 dwords
    unsigned int* W2p = W1p + 16384;                             // 64*128/2  = 4096 dwords

    const int mtiles  = (N + 15) / 16;
    const int eblocks = (E + 7) / 8;       // 8 waves (edges) per 256-thread block
    const int lsblks  = (N + 7) / 8;       // 8 rows per 256-thread block

    // One-time weight pack (f32 -> f16, pre-swizzled to B-fragment layout).
    pack_weights<256, 128><<<(16384 + 255) / 256, 256, 0, stream>>>(W1, W1p);
    pack_weights<128, 64><<<(4096 + 255) / 256, 256, 0, stream>>>(W2, W2p);

    // Layer 1: h1 = x @ W1^T + b1 ; agg1 = scatter_add(h1[src] -> dst)
    zero_kernel<<<1024, 256, 0, stream>>>(agg, N * 128);
    gemm_wmma<256, 128, false><<<mtiles, 256, 0, stream>>>(x, W1p, b1, h, N);
    scatter_add<128><<<eblocks, 256, 0, stream>>>(h, src, dst, agg, E);

    // Layer 2: h2 = relu(agg1) @ W2^T + b2 (relu fused into A staging)
    gemm_wmma<128, 64, true><<<mtiles, 128, 0, stream>>>(agg, W2p, b2, h, N);
    zero_kernel<<<1024, 256, 0, stream>>>(agg, N * 64);    // agg reused as agg2
    scatter_add<64><<<eblocks, 256, 0, stream>>>(h, src, dst, agg, E);

    // log_softmax over 64 classes per node
    log_softmax64<<<lsblks, 256, 0, stream>>>(agg, out, N);
}
